// MotionMTA_80461917323390
// MI455X (gfx1250) — matrix operand order repaired and checked by reading, not compiled, not run on hardware
//
#include <hip/hip_runtime.h>

// ---------------------------------------------------------------------------
// Types for CDNA5 WMMA (wave32, bf16 -> f32 accumulate)
// ---------------------------------------------------------------------------
typedef __bf16 bf16;
typedef bf16 v16bf __attribute__((ext_vector_type(16)));
typedef float v8f __attribute__((ext_vector_type(8)));
typedef unsigned int u32x4 __attribute__((ext_vector_type(4)));
typedef float f32x4 __attribute__((ext_vector_type(4)));

static_assert(sizeof(v16bf) == 32, "v16bf size");

// ---------------------------------------------------------------------------
// Problem constants
// ---------------------------------------------------------------------------
constexpr int Bc = 64, Tn = 240, Dc = 512, TDc = 768, EDc = 2048;
constexpr int Hc = 8, HDc = 64, Nc = 77, D2c = 1024, T3c = 80;
constexpr long long SZ = (long long)Bc * Tn * Dc;   // 7,864,320
constexpr int Z2 = Bc * Hc;                         // 512

// ---------------------------------------------------------------------------
// Generic batched GEMM: C[z][m][n] = alpha * sum_k A[z][m][k]*B[z][n][k]
//                                    (+Cold) (+bias[n]) (+res)
// aL/bL = 0: operand stored (rows=M/N, cols=K) row-major, row stride *_rs.
//            Constraint: K % 8 == 0 (true at all aL/bL==0 call sites).
// aL/bL = 1: operand stored (rows=K, cols=M/N) row-major, row stride *_rs.
//            Constraint: Cols % 8 == 0 (true at all call sites).
// Out-of-range M/N rows are CLAMPED while staging (their outputs are never
// stored); out-of-range K segments are ZERO-filled (uniform per thread).
// All row strides at every call site are multiples of 4 floats (16B).
// ---------------------------------------------------------------------------
struct GemmP {
  const float* A; const float* Bw; float* C; const float* bias;
  const float* res; const float* alpha;
  long long a_sb, a_sh, b_sb, b_sh, c_sb, c_sh, r_sb, r_sh;
  int a_rs, b_rs, c_rs, r_rs;
  int M, N, K, ZH, aL, bL, acc, amod;
};

// stage one 64(row)x32(k) tile of a row-major (rows,K) operand into LDS
__device__ __forceinline__ void stage_rk(const float* __restrict__ base, bf16* lds,
                                         int row_base, int Rows, int k0, int K,
                                         int rs, int tid) {
  const int row = tid >> 2, seg = (tid & 3) * 8;
  const int rg = min(row_base + row, Rows - 1);        // clamp: OOB rows unused
  u32x4* dst = reinterpret_cast<u32x4*>(&lds[row * 40 + seg]);
  if (k0 + seg + 8 <= K) {                             // uniform per thread
    const float* src = base + (long long)rg * rs + k0 + seg;
    const f32x4 v0 = *reinterpret_cast<const f32x4*>(src);
    const f32x4 v1 = *reinterpret_cast<const f32x4*>(src + 4);
    union { bf16 h[8]; u32x4 q; } pk;
#pragma unroll
    for (int i = 0; i < 4; ++i) pk.h[i] = (bf16)v0[i];
#pragma unroll
    for (int i = 0; i < 4; ++i) pk.h[4 + i] = (bf16)v1[i];
    *dst = pk.q;
  } else {
    *dst = (u32x4){0u, 0u, 0u, 0u};                    // zero K-tail
  }
}

// stage one tile of a K-major (K, cols) operand into LDS[col][k] (transposed)
__device__ __forceinline__ void stage_km(const float* __restrict__ base, bf16* lds,
                                         int col_base, int Cols, int k0, int K,
                                         int rs, int tid) {
  const int kr = tid >> 3, seg = (tid & 7) * 8;
  const int kg = k0 + kr;
  // uniform per thread: k row valid AND 8-col segment fully inside Cols
  if (kg < K && col_base + seg + 8 <= Cols) {
    const float* src = base + (long long)kg * rs + col_base + seg;
    const f32x4 v0 = *reinterpret_cast<const f32x4*>(src);
    const f32x4 v1 = *reinterpret_cast<const f32x4*>(src + 4);
#pragma unroll
    for (int i = 0; i < 4; ++i) lds[(seg + i) * 40 + kr] = (bf16)v0[i];
#pragma unroll
    for (int i = 0; i < 4; ++i) lds[(seg + 4 + i) * 40 + kr] = (bf16)v1[i];
  } else {
#pragma unroll
    for (int i = 0; i < 8; ++i) lds[(seg + i) * 40 + kr] = (bf16)0.0f;
  }
}

__global__ __launch_bounds__(256) void k_gemm(GemmP p) {
  __shared__ __align__(16) bf16 As[64 * 40];
  __shared__ __align__(16) bf16 Bs[64 * 40];
  const int z  = blockIdx.z;
  const int zb = z / p.ZH, zh = z % p.ZH;
  const float* __restrict__ Ab = p.A  + (long long)zb * p.a_sb + (long long)zh * p.a_sh;
  const float* __restrict__ Bb = p.Bw + (long long)zb * p.b_sb + (long long)zh * p.b_sh;
  const int m_base = blockIdx.y * 64;
  const int n_base = blockIdx.x * 64;
  const int tid  = threadIdx.x;
  const int wave = tid >> 5, lane = tid & 31;
  const int wr = wave >> 1;     // 0..3 : 16-row strip
  const int wc = wave & 1;      // 0..1 : 32-col strip
  const int lm = lane & 15, half = lane >> 4;

  v8f acc0 = {0.f,0.f,0.f,0.f,0.f,0.f,0.f,0.f};
  v8f acc1 = acc0;

  for (int k0 = 0; k0 < p.K; k0 += 32) {
    if (p.aL == 0) stage_rk(Ab, As, m_base, p.M, k0, p.K, p.a_rs, tid);
    else           stage_km(Ab, As, m_base, p.M, k0, p.K, p.a_rs, tid);
    if (p.bL == 0) stage_rk(Bb, Bs, n_base, p.N, k0, p.K, p.b_rs, tid);
    else           stage_km(Bb, Bs, n_base, p.N, k0, p.K, p.b_rs, tid);
    __syncthreads();

    // ---- fragments (ds_load_b128 each) ----
    union Frag { v16bf v; u32x4 q[2]; };
    Frag fa, fb0, fb1;
    const int am = wr * 16 + lm;
    // A: K = half*16 + 2v
    fa.q[0] = *reinterpret_cast<const u32x4*>(&As[am * 40 + half * 16]);
    fa.q[1] = *reinterpret_cast<const u32x4*>(&As[am * 40 + half * 16 + 8]);
    const int bn0 = wc * 32 + lm;
    // B: K = half*16 + 2v
    fb0.q[0] = *reinterpret_cast<const u32x4*>(&Bs[bn0 * 40 + half * 16]);
    fb0.q[1] = *reinterpret_cast<const u32x4*>(&Bs[bn0 * 40 + half * 16 + 8]);
    const int bn1 = bn0 + 16;
    fb1.q[0] = *reinterpret_cast<const u32x4*>(&Bs[bn1 * 40 + half * 16]);
    fb1.q[1] = *reinterpret_cast<const u32x4*>(&Bs[bn1 * 40 + half * 16 + 8]);

    acc0 = __builtin_amdgcn_wmma_f32_16x16x32_bf16(false, fa.v, false, fb0.v,
                                                   (short)0, acc0, false, false);
    acc1 = __builtin_amdgcn_wmma_f32_16x16x32_bf16(false, fa.v, false, fb1.v,
                                                   (short)0, acc1, false, false);
    __syncthreads();
  }

  // ---------------- epilogue ----------------
  const float alpha = p.alpha ? p.alpha[p.amod > 0 ? (z % p.amod) : 0] : 1.0f;
  float* Cb = p.C + (long long)zb * p.c_sb + (long long)zh * p.c_sh;
  const float* Rb = p.res ? (p.res + (long long)zb * p.r_sb + (long long)zh * p.r_sh)
                          : (const float*)nullptr;
  const int ng0 = n_base + wc * 32 + lm;   // column of acc0
  const int ng1 = ng0 + 16;                // column of acc1
  const int mg0 = m_base + wr * 16 + half * 8;  // rows mg0..mg0+7

  float vv[16];
#pragma unroll
  for (int r = 0; r < 8; ++r) { vv[r] = acc0[r] * alpha; vv[8 + r] = acc1[r] * alpha; }

  if (m_base + 64 <= p.M && n_base + 64 <= p.N) {
    if (p.acc) {
#pragma unroll
      for (int r = 0; r < 8; ++r) {
        vv[r]     += Cb[(long long)(mg0 + r) * p.c_rs + ng0];
        vv[8 + r] += Cb[(long long)(mg0 + r) * p.c_rs + ng1];
      }
    }
    if (p.bias) {
      const float b0 = p.bias[ng0], b1 = p.bias[ng1];
#pragma unroll
      for (int r = 0; r < 8; ++r) { vv[r] += b0; vv[8 + r] += b1; }
    }
    if (Rb) {
#pragma unroll
      for (int r = 0; r < 8; ++r) {
        vv[r]     += Rb[(long long)(mg0 + r) * p.r_rs + ng0];
        vv[8 + r] += Rb[(long long)(mg0 + r) * p.r_rs + ng1];
      }
    }
#pragma unroll
    for (int r = 0; r < 8; ++r) {
      Cb[(long long)(mg0 + r) * p.c_rs + ng0] = vv[r];
      Cb[(long long)(mg0 + r) * p.c_rs + ng1] = vv[8 + r];
    }
  } else {
    // edge tile: guarded scalar stores
#pragma unroll
    for (int t = 0; t < 2; ++t) {
      const int ng = t ? ng1 : ng0;
      if (ng < p.N) {
#pragma unroll
        for (int r = 0; r < 8; ++r) {
          const int mg = mg0 + r;
          if (mg < p.M) {
            const long long ci = (long long)mg * p.c_rs + ng;
            float v = vv[t * 8 + r];
            if (p.acc)  v += Cb[ci];
            if (p.bias) v += p.bias[ng];
            if (Rb)     v += Rb[(long long)mg * p.r_rs + ng];
            Cb[ci] = v;
          }
        }
      }
    }
  }
}

// ---------------------------------------------------------------------------
// Support kernels
// ---------------------------------------------------------------------------
__global__ __launch_bounds__(256) void k_layernorm(const float* __restrict__ x,
                                                   const float* __restrict__ g,
                                                   const float* __restrict__ b,
                                                   float* __restrict__ y, int width) {
  __shared__ float red[256];
  const long long row = blockIdx.x;
  const float* xr = x + row * width;
  float s = 0.f;
  for (int i = threadIdx.x; i < width; i += 256) s += xr[i];
  red[threadIdx.x] = s; __syncthreads();
  for (int o = 128; o > 0; o >>= 1) { if (threadIdx.x < o) red[threadIdx.x] += red[threadIdx.x + o]; __syncthreads(); }
  const float mean = red[0] / width; __syncthreads();
  float v = 0.f;
  for (int i = threadIdx.x; i < width; i += 256) { float d = xr[i] - mean; v += d * d; }
  red[threadIdx.x] = v; __syncthreads();
  for (int o = 128; o > 0; o >>= 1) { if (threadIdx.x < o) red[threadIdx.x] += red[threadIdx.x + o]; __syncthreads(); }
  const float rstd = rsqrtf(red[0] / width + 1e-5f);
  float* yr = y + row * width;
  for (int i = threadIdx.x; i < width; i += 256) yr[i] = (xr[i] - mean) * rstd * g[i] + b[i];
}

__global__ __launch_bounds__(256) void k_softmax(float* __restrict__ x, int width) {
  __shared__ float red[256];
  float* xr = x + (long long)blockIdx.x * width;
  float mx = -1e30f;
  for (int i = threadIdx.x; i < width; i += 256) mx = fmaxf(mx, xr[i]);
  red[threadIdx.x] = mx; __syncthreads();
  for (int o = 128; o > 0; o >>= 1) { if (threadIdx.x < o) red[threadIdx.x] = fmaxf(red[threadIdx.x], red[threadIdx.x + o]); __syncthreads(); }
  const float m = red[0]; __syncthreads();
  float s = 0.f;
  for (int i = threadIdx.x; i < width; i += 256) { float e = expf(xr[i] - m); xr[i] = e; s += e; }
  red[threadIdx.x] = s; __syncthreads();
  for (int o = 128; o > 0; o >>= 1) { if (threadIdx.x < o) red[threadIdx.x] += red[threadIdx.x + o]; __syncthreads(); }
  const float inv = 1.0f / red[0];
  for (int i = threadIdx.x; i < width; i += 256) xr[i] *= inv;
}

__global__ void k_transpose(const float* __restrict__ in, float* __restrict__ out,
                            int Rr, int Cc, long long n) {
  long long i = (long long)blockIdx.x * 256 + threadIdx.x;
  if (i >= n) return;
  int c = (int)(i % Cc); long long t = i / Cc;
  int r = (int)(t % Rr); long long z = t / Rr;
  out[(z * Cc + c) * Rr + r] = in[i];
}

__global__ void k_silu(const float* __restrict__ in, float* __restrict__ out, long long n) {
  long long i = (long long)blockIdx.x * 256 + threadIdx.x;
  if (i >= n) return;
  float v = in[i]; out[i] = v / (1.0f + expf(-v));
}

__global__ void k_gelu(float* __restrict__ x, long long n) {   // exact erf GELU, in place
  long long i = (long long)blockIdx.x * 256 + threadIdx.x;
  if (i >= n) return;
  float v = x[i];
  x[i] = 0.5f * v * (1.0f + erff(v * 0.70710678118654752f));
}

__global__ void k_add(const float* __restrict__ a, const float* __restrict__ b,
                      float* __restrict__ out, long long n) {
  long long i = (long long)blockIdx.x * 256 + threadIdx.x;
  if (i >= n) return;
  out[i] = a[i] + b[i];
}

// out = silu( ln * (1 + eo[b, c]) + eo[b, 512 + c] )
__global__ void k_ssf(const float* __restrict__ ln, const float* __restrict__ eo,
                      float* __restrict__ out, long long n) {
  long long i = (long long)blockIdx.x * 256 + threadIdx.x;
  if (i >= n) return;
  int c = (int)(i % Dc);
  long long bt = i / Dc;
  int b = (int)(bt / Tn);
  float sc = eo[(long long)b * D2c + c];
  float sh = eo[(long long)b * D2c + Dc + c];
  float v = ln[i] * (1.0f + sc) + sh;
  out[i] = v / (1.0f + expf(-v));
}

// dst[b, chunk*80 + t, :] = src[b, t, :] + xf_[b, chunk*80 + t, :]
__global__ void k_ccat(const float* __restrict__ src, const float* __restrict__ xf_,
                       float* __restrict__ dst, int chunk, long long n) {
  long long i = (long long)blockIdx.x * 256 + threadIdx.x;
  if (i >= n) return;
  int c = (int)(i % Dc);
  long long r = i / Dc;
  int t = (int)(r % T3c);
  int b = (int)(r / T3c);
  long long di = ((long long)b * Tn + chunk * T3c + t) * Dc + c;
  dst[di] = src[i] + xf_[di];
}

// out = l1*softmax0 + l2*softmax1 over the stacked axis
__global__ void k_fusion(const float* __restrict__ l1, const float* __restrict__ l2,
                         const float* __restrict__ f1, const float* __restrict__ f2,
                         float* __restrict__ out, long long n) {
  long long i = (long long)blockIdx.x * 256 + threadIdx.x;
  if (i >= n) return;
  float a = f1[i], b = f2[i];
  float m = fmaxf(a, b);
  float e1 = expf(a - m), e2 = expf(b - m);
  float inv = 1.0f / (e1 + e2);
  out[i] = l1[i] * (e1 * inv) + l2[i] * (e2 * inv);
}

// zero-pad tokens: in (B,T,512) -> out (B,T+2,512)
__global__ void k_pad(const float* __restrict__ in, float* __restrict__ out, long long n) {
  long long i = (long long)blockIdx.x * 256 + threadIdx.x;
  if (i >= n) return;
  int c = (int)(i % Dc);
  long long r = i / Dc;
  int t = (int)(r % (Tn + 2));
  int b = (int)(r / (Tn + 2));
  out[i] = (t == 0 || t == Tn + 1) ? 0.0f : in[((long long)b * Tn + (t - 1)) * Dc + c];
}

// conv weight (O,I,3) -> 3 contiguous (O,I) matrices
__global__ void k_repack(const float* __restrict__ w, float* __restrict__ out, long long n) {
  long long i = (long long)blockIdx.x * 256 + threadIdx.x;
  if (i >= n) return;
  int kk = (int)(i / ((long long)Dc * Dc));
  long long rem = i % ((long long)Dc * Dc);
  int o = (int)(rem / Dc), ii = (int)(rem % Dc);
  out[i] = w[((long long)o * Dc + ii) * 3 + kk];
}

// DCT_M[k,n] = 2 cos(pi (2n+1) k / (2D));  IDCT_M[n,k] = cos(pi (2n+1) k / (2D)) * w_k / D
__global__ void k_dct(float* __restrict__ dct, float* __restrict__ idct) {
  long long i = (long long)blockIdx.x * 256 + threadIdx.x;
  if (i >= (long long)Dc * Dc) return;
  int r = (int)(i / Dc), c = (int)(i % Dc);
  const double pi = 3.14159265358979323846;
  dct[i]  = (float)(2.0 * cos(pi * (2.0 * c + 1.0) * r / (2.0 * Dc)));
  double w = (c == 0) ? 0.5 : 1.0;
  idct[i] = (float)(cos(pi * (2.0 * r + 1.0) * c / (2.0 * Dc)) * w / Dc);
}

// ---------------------------------------------------------------------------
// Host-side helpers
// ---------------------------------------------------------------------------
static inline dim3 ew(long long n) { return dim3((unsigned)((n + 255) / 256)); }

static void run_gemm(hipStream_t s, GemmP p, int Z) {
  dim3 g((unsigned)((p.N + 63) / 64), (unsigned)((p.M + 63) / 64), (unsigned)Z);
  k_gemm<<<g, dim3(256), 0, s>>>(p);
}

// flat NT GEMM:  C (M,N) = A (M,K) @ W(N,K)^T + bias + res
static void gemm_nt(hipStream_t s, const float* A, const float* Wt, const float* bias,
                    float* C, int M, int N, int K, const float* res) {
  GemmP p{};
  p.A = A; p.a_rs = K;
  p.Bw = Wt; p.b_rs = K;
  p.C = C; p.c_rs = N;
  p.bias = bias;
  p.res = res; p.r_rs = N;
  p.M = M; p.N = N; p.K = K; p.ZH = 1;
  run_gemm(s, p, 1);
}

// Standard MHSA (as in reference mhsa()): raw-reshape head views.
static void mhsa_run(hipStream_t s, const float* x, long long x_off, int Tcur,
                     const float* const* Wp, float* out,
                     float* qb, float* kb, float* vb, float* ktb, float* attb) {
  // q,k,v: (B, Tcur, 512)
  for (int j = 0; j < 3; ++j) {
    GemmP p{};
    p.A = x + x_off; p.a_sb = (long long)Tn * Dc; p.a_rs = Dc;
    p.Bw = Wp[2 * j]; p.b_rs = Dc;
    p.bias = Wp[2 * j + 1];
    p.C = (j == 0) ? qb : (j == 1) ? kb : vb;
    p.c_sb = (long long)Tcur * Dc; p.c_rs = Dc;
    p.M = Tcur; p.N = Dc; p.K = Dc; p.ZH = 1;
    run_gemm(s, p, Bc);
  }
  // kt: (Z2, hd, Tcur) -> (Z2, Tcur, hd)
  {
    long long n = (long long)Z2 * HDc * Tcur;
    k_transpose<<<ew(n), 256, 0, s>>>(kb, ktb, HDc, Tcur, n);
  }
  // scores[t,s] = temp * sum_d k[d,t] v[d,s]
  {
    GemmP p{};
    p.A = ktb; p.a_sb = (long long)Tcur * HDc; p.a_rs = HDc;
    p.Bw = vb; p.b_sb = (long long)HDc * Tcur; p.b_rs = Tcur; p.bL = 1;
    p.C = attb; p.c_sb = (long long)Tcur * Tcur; p.c_rs = Tcur;
    p.M = Tcur; p.N = Tcur; p.K = HDc; p.ZH = 1;
    p.alpha = Wp[6]; p.amod = Hc;
    run_gemm(s, p, Z2);
  }
  k_softmax<<<dim3((unsigned)(Z2 * Tcur)), 256, 0, s>>>(attb, Tcur);
  // out[d,s] = sum_t q[d,t] attn[t,s]  (writes raw-reshaped (B,Tcur,512))
  {
    GemmP p{};
    p.A = qb; p.a_sb = (long long)HDc * Tcur; p.a_rs = Tcur;
    p.Bw = attb; p.b_sb = (long long)Tcur * Tcur; p.b_rs = Tcur; p.bL = 1;
    p.C = out; p.c_sb = (long long)HDc * Tcur; p.c_rs = Tcur;
    p.M = HDc; p.N = Tcur; p.K = Tcur; p.ZH = 1;
    run_gemm(s, p, Z2);
  }
}

// Motion MHSA (mhsa_m): scores over head dims, output + residual.
static void mhsa_m_run(hipStream_t s, const float* xln, const float* const* Wp,
                       const float* res, float* out,
                       float* qb, float* kb, float* vb, float* sb) {
  gemm_nt(s, xln, Wp[0], Wp[1], qb, Bc * Tn, Dc, Dc, nullptr);
  gemm_nt(s, xln, Wp[2], Wp[3], kb, Bc * Tn, Dc, Dc, nullptr);
  gemm_nt(s, xln, Wp[4], Wp[5], vb, Bc * Tn, Dc, Dc, nullptr);
  // S[d,e] = temp * sum_t q[d,t] k[e,t]
  {
    GemmP p{};
    p.A = qb; p.a_sb = (long long)HDc * Tn; p.a_rs = Tn;
    p.Bw = kb; p.b_sb = (long long)HDc * Tn; p.b_rs = Tn;           // NT
    p.C = sb; p.c_sb = (long long)HDc * HDc; p.c_rs = HDc;
    p.M = HDc; p.N = HDc; p.K = Tn; p.ZH = 1;
    p.alpha = Wp[6]; p.amod = Hc;
    run_gemm(s, p, Z2);
  }
  k_softmax<<<dim3((unsigned)(Z2 * HDc)), 256, 0, s>>>(sb, HDc);
  // O[d,t] = sum_e S[d,e] v[e,t] + res
  {
    GemmP p{};
    p.A = sb; p.a_sb = (long long)HDc * HDc; p.a_rs = HDc;
    p.Bw = vb; p.b_sb = (long long)HDc * Tn; p.b_rs = Tn; p.bL = 1;
    p.C = out; p.c_sb = (long long)HDc * Tn; p.c_rs = Tn;
    p.res = res; p.r_sb = (long long)HDc * Tn; p.r_rs = Tn;
    p.M = HDc; p.N = Tn; p.K = HDc; p.ZH = 1;
    run_gemm(s, p, Z2);
  }
}

// StylizationBlock: out = res + W_out @ silu( ln(h) * (1+scale) + shift )
static void styl_run(hipStream_t s, const float* h, const float* const* Wp /*ew,eb,ow,ob,lg,lb*/,
                     const float* E, const float* res, float* out,
                     float* eo, float* lnbuf, float* sbuf) {
  {
    GemmP p{};
    p.A = E; p.a_rs = EDc;
    p.Bw = Wp[0]; p.b_rs = EDc; p.bias = Wp[1];
    p.C = eo; p.c_rs = D2c;
    p.M = Bc; p.N = D2c; p.K = EDc; p.ZH = 1;
    run_gemm(s, p, 1);
  }
  k_layernorm<<<dim3((unsigned)(Bc * Tn)), 256, 0, s>>>(h, Wp[4], Wp[5], lnbuf, Dc);
  k_ssf<<<ew(SZ), 256, 0, s>>>(lnbuf, eo, sbuf, SZ);
  gemm_nt(s, sbuf, Wp[2], Wp[3], out, Bc * Tn, Dc, Dc, res);
}

// 3-tap conv over tokens = 3 shifted GEMMs on padded input, accumulate in C.
static void conv3(hipStream_t s, const float* pad, const float* Wk, const float* bias,
                  float* out, const float* res) {
  for (int k = 0; k < 3; ++k) {
    GemmP p{};
    p.A = pad + k * Dc; p.a_sb = (long long)(Tn + 2) * Dc; p.a_rs = Dc;
    p.Bw = Wk + (long long)k * Dc * Dc; p.b_rs = Dc;
    p.C = out; p.c_sb = (long long)Tn * Dc; p.c_rs = Dc;
    p.M = Tn; p.N = Dc; p.K = Dc; p.ZH = 1;
    p.bias = (k == 0) ? bias : nullptr;
    p.acc = (k > 0);
    if (k == 2 && res) { p.res = res; p.r_sb = (long long)Tn * Dc; p.r_rs = Dc; }
    run_gemm(s, p, Bc);
  }
}

// ---------------------------------------------------------------------------
// Full forward
// ---------------------------------------------------------------------------
extern "C" void kernel_launch(void* const* d_in, const int* in_sizes, int n_in,
                              void* d_out, int out_size, void* d_ws, size_t ws_size,
                              hipStream_t stream) {
  (void)in_sizes; (void)n_in; (void)out_size; (void)ws_size;
  auto Pp = [&](int i) -> const float* { return (const float*)d_in[i]; };

  // --- workspace bump allocator (floats) ---
  float* Wb = (float*)d_ws;
  long long off = 0;
  auto alloc = [&](long long n) -> float* { float* p = Wb + off; off += (n + 255) & ~255LL; return p; };

  float* F_DCT  = alloc((long long)Dc * Dc);
  float* F_IDCT = alloc((long long)Dc * Dc);
  float* F_CW1  = alloc(3LL * Dc * Dc);
  float* F_CW2  = alloc(3LL * Dc * Dc);
  float* F_E    = alloc((long long)Bc * EDc);
  float* F_EO   = alloc((long long)Bc * D2c);
  float* F_LNB  = alloc(SZ);
  float* F_S    = alloc(SZ);
  float* F_Q    = alloc(SZ);
  float* F_K    = alloc(SZ);
  float* F_V    = alloc(SZ);
  float* F_KT   = alloc(SZ);
  float* F_C    = alloc(SZ);
  float* F_X0   = alloc(SZ);
  float* F_Y    = alloc(SZ);
  float* F_XB   = alloc(SZ);
  float* F_L1   = alloc(SZ);
  float* F_L2   = alloc(SZ);
  float* F_A    = alloc(SZ);
  float* F_TA   = alloc(SZ);
  float* F_YM   = alloc(SZ);
  float* F_Y2   = alloc(SZ);
  float* F_MA0  = alloc(SZ);
  float* F_MAB  = alloc(SZ);
  float* F_ATT  = alloc((long long)Z2 * Tn * Tn);
  float* F_H1   = alloc((long long)Bc * Tn * D2c);
  float* F_TX   = alloc((long long)Bc * Nc * TDc);
  float* F_KC   = alloc((long long)Bc * Nc * Dc);
  float* F_VC   = alloc((long long)Bc * Nc * Dc);
  float* F_AS   = alloc((long long)Z2 * HDc * HDc);
  float* F_P1   = alloc((long long)Bc * (Tn + 2) * Dc);

  const float* x   = Pp(0);
  const float* xf  = Pp(1);
  const float* emb = Pp(2);
  const long long NH1 = (long long)Bc * Tn * D2c;

  // --- constants / repacks ---
  k_dct<<<ew((long long)Dc * Dc), 256, 0, stream>>>(F_DCT, F_IDCT);
  k_repack<<<ew(3LL * Dc * Dc), 256, 0, stream>>>(Pp(89), F_CW1, 3LL * Dc * Dc);
  k_repack<<<ew(3LL * Dc * Dc), 256, 0, stream>>>(Pp(91), F_CW2, 3LL * Dc * Dc);
  k_silu<<<ew((long long)Bc * EDc), 256, 0, stream>>>(emb, F_E, (long long)Bc * EDc);

  // --- y0 = lnorm(x, P.ln1) ---
  k_layernorm<<<dim3((unsigned)(Bc * Tn)), 256, 0, stream>>>(x, Pp(3), Pp(4), F_X0, Dc);

  // ================= cross_attn(y0, xf, emb) -> F_Y =================
  k_layernorm<<<dim3((unsigned)(Bc * Tn)), 256, 0, stream>>>(F_X0, Pp(13), Pp(14), F_LNB, Dc);
  gemm_nt(stream, F_LNB, Pp(7), Pp(8), F_Q, Bc * Tn, Dc, Dc, nullptr);          // q
  k_softmax<<<dim3((unsigned)(Bc * Tn * Hc)), 256, 0, stream>>>(F_Q, HDc);       // softmax over hd
  k_layernorm<<<dim3((unsigned)(Bc * Nc)), 256, 0, stream>>>(xf, Pp(15), Pp(16), F_TX, TDc);
  gemm_nt(stream, F_TX, Pp(9), Pp(10), F_KC, Bc * Nc, Dc, TDc, nullptr);         // k
  k_softmax<<<dim3((unsigned)(Bc * Nc * Hc)), 256, 0, stream>>>(F_KC, HDc);
  gemm_nt(stream, F_TX, Pp(11), Pp(12), F_VC, Bc * Nc, Dc, TDc, nullptr);        // v
  {  // Att[d,l] = sum_n K[n,d] V[n,l]   (per b,h)
    GemmP p{};
    p.A = F_KC; p.a_sb = (long long)Nc * Dc; p.a_sh = HDc; p.a_rs = Dc; p.aL = 1;
    p.Bw = F_VC; p.b_sb = (long long)Nc * Dc; p.b_sh = HDc; p.b_rs = Dc; p.bL = 1;
    p.C = F_AS; p.c_sb = (long long)Hc * HDc * HDc; p.c_sh = (long long)HDc * HDc; p.c_rs = HDc;
    p.M = HDc; p.N = HDc; p.K = Nc; p.ZH = Hc;
    run_gemm(stream, p, Z2);
  }
  {  // out[t,l] = sum_d q[t,d] Att[d,l]
    GemmP p{};
    p.A = F_Q; p.a_sb = (long long)Tn * Dc; p.a_sh = HDc; p.a_rs = Dc;
    p.Bw = F_AS; p.b_sb = (long long)Hc * HDc * HDc; p.b_sh = (long long)HDc * HDc; p.b_rs = HDc; p.bL = 1;
    p.C = F_C; p.c_sb = (long long)Tn * Dc; p.c_sh = HDc; p.c_rs = Dc;
    p.M = Tn; p.N = HDc; p.K = HDc; p.ZH = Hc;
    run_gemm(stream, p, Z2);
  }
  {
    const float* Wst[6] = {Pp(17), Pp(18), Pp(19), Pp(20), Pp(21), Pp(22)};
    styl_run(stream, F_C, Wst, F_E, F_X0, F_Y, F_EO, F_LNB, F_S);
  }

  // ================= temporal_attn(F_Y) -> F_A =================
  const float* A0[7]  = {Pp(23), Pp(24), Pp(25), Pp(26), Pp(27), Pp(28), Pp(29)};
  const float* L11[7] = {Pp(30), Pp(31), Pp(32), Pp(33), Pp(34), Pp(35), Pp(36)};
  const float* L12[7] = {Pp(37), Pp(38), Pp(39), Pp(40), Pp(41), Pp(42), Pp(43)};
  const float* L13[7] = {Pp(44), Pp(45), Pp(46), Pp(47), Pp(48), Pp(49), Pp(50)};
  const float* L21[7] = {Pp(51), Pp(52), Pp(53), Pp(54), Pp(55), Pp(56), Pp(57)};

  mhsa_run(stream, F_Y, 0, Tn, A0, F_XB, F_Q, F_K, F_V, F_KT, F_ATT);            // x_
  for (int c = 0; c < 3; ++c) {
    const float* const* Wc = (c == 0) ? L11 : (c == 1) ? L12 : L13;
    mhsa_run(stream, F_Y, (long long)c * T3c * Dc, T3c, Wc, F_C, F_Q, F_K, F_V, F_KT, F_ATT);
    k_ccat<<<ew((long long)Bc * T3c * Dc), 256, 0, stream>>>(F_C, F_XB, F_L1, c,
                                                             (long long)Bc * T3c * Dc);
  }
  k_layernorm<<<dim3((unsigned)(Bc * Tn)), 256, 0, stream>>>(F_L1, Pp(58), Pp(59), F_LNB, Dc);
  mhsa_run(stream, F_LNB, 0, Tn, L21, F_C, F_Q, F_K, F_V, F_KT, F_ATT);
  k_add<<<ew(SZ), 256, 0, stream>>>(F_C, F_XB, F_L2, SZ);
  gemm_nt(stream, F_L1, Pp(60), Pp(61), F_Q, Bc * Tn, Dc, Dc, nullptr);          // fusion(l1)
  gemm_nt(stream, F_L2, Pp(60), Pp(61), F_K, Bc * Tn, Dc, Dc, nullptr);          // fusion(l2)
  k_fusion<<<ew(SZ), 256, 0, stream>>>(F_L1, F_L2, F_Q, F_K, F_A, SZ);
  {
    const float* Wst[6] = {Pp(97), Pp(98), Pp(99), Pp(100), Pp(101), Pp(102)};
    styl_run(stream, F_A, Wst, F_E, F_Y, F_TA, F_EO, F_LNB, F_S);                // t_a
  }

  // ================= motion_attn(F_TA) -> F_MA0 =================
  const float* MA[7] = {Pp(62), Pp(63), Pp(64), Pp(65), Pp(66), Pp(67), Pp(68)};
  k_layernorm<<<dim3((unsigned)(Bc * Tn)), 256, 0, stream>>>(F_TA, Pp(69), Pp(70), F_LNB, Dc);
  mhsa_m_run(stream, F_LNB, MA, F_TA, F_YM, F_Q, F_K, F_V, F_AS);                // y
  // DCT path
  k_layernorm<<<dim3((unsigned)(Bc * Tn)), 256, 0, stream>>>(F_YM, Pp(71), Pp(72), F_LNB, Dc);
  gemm_nt(stream, F_LNB, F_DCT, nullptr, F_C, Bc * Tn, Dc, Dc, nullptr);         // d = ln @ DCT^T
  gemm_nt(stream, F_C, Pp(77), Pp(78), F_H1, Bc * Tn, D2c, Dc, nullptr);         // lin1
  k_gelu<<<ew(NH1), 256, 0, stream>>>(F_H1, NH1);
  gemm_nt(stream, F_H1, Pp(79), Pp(80), F_Q, Bc * Tn, Dc, D2c, nullptr);         // lin2
  gemm_nt(stream, F_Q, F_IDCT, nullptr, F_C, Bc * Tn, Dc, Dc, nullptr);          // @ IDCT^T
  // m1 = mlp1(d) + y
  gemm_nt(stream, F_C, Pp(81), Pp(82), F_H1, Bc * Tn, D2c, Dc, nullptr);
  k_gelu<<<ew(NH1), 256, 0, stream>>>(F_H1, NH1);
  gemm_nt(stream, F_H1, Pp(83), Pp(84), F_K, Bc * Tn, Dc, D2c, F_YM);
  // m2 = mlp2(ln3(y)) + m1 -> y2
  k_layernorm<<<dim3((unsigned)(Bc * Tn)), 256, 0, stream>>>(F_YM, Pp(73), Pp(74), F_LNB, Dc);
  gemm_nt(stream, F_LNB, Pp(85), Pp(86), F_H1, Bc * Tn, D2c, Dc, nullptr);
  k_gelu<<<ew(NH1), 256, 0, stream>>>(F_H1, NH1);
  gemm_nt(stream, F_H1, Pp(87), Pp(88), F_Y2, Bc * Tn, Dc, D2c, F_K);
  // conv block
  k_layernorm<<<dim3((unsigned)(Bc * Tn)), 256, 0, stream>>>(F_Y2, Pp(75), Pp(76), F_LNB, Dc);
  k_pad<<<ew((long long)Bc * (Tn + 2) * Dc), 256, 0, stream>>>(F_LNB, F_P1, (long long)Bc * (Tn + 2) * Dc);
  conv3(stream, F_P1, F_CW1, Pp(90), F_C, nullptr);                              // conv1
  k_silu<<<ew(SZ), 256, 0, stream>>>(F_C, F_C, SZ);                              // silu
  k_pad<<<ew((long long)Bc * (Tn + 2) * Dc), 256, 0, stream>>>(F_C, F_P1, (long long)Bc * (Tn + 2) * Dc);
  conv3(stream, F_P1, F_CW2, Pp(92), F_MA0, F_Y2);                               // conv2 + y2
  {
    const float* Wst[6] = {Pp(103), Pp(104), Pp(105), Pp(106), Pp(107), Pp(108)};
    styl_run(stream, F_MA0, Wst, F_E, F_TA, F_MAB, F_EO, F_LNB, F_S);            // m_a
  }

  // ================= final: styl(mlp(ln2(m_a))) + m_a -> d_out =================
  k_layernorm<<<dim3((unsigned)(Bc * Tn)), 256, 0, stream>>>(F_MAB, Pp(5), Pp(6), F_LNB, Dc);
  gemm_nt(stream, F_LNB, Pp(93), Pp(94), F_H1, Bc * Tn, D2c, Dc, nullptr);
  k_gelu<<<ew(NH1), 256, 0, stream>>>(F_H1, NH1);
  gemm_nt(stream, F_H1, Pp(95), Pp(96), F_C, Bc * Tn, Dc, D2c, nullptr);
  {
    const float* Wst[6] = {Pp(109), Pp(110), Pp(111), Pp(112), Pp(113), Pp(114)};
    styl_run(stream, F_C, Wst, F_E, F_MAB, (float*)d_out, F_EO, F_LNB, F_S);
  }
}
